// Attention_61211873902675
// MI455X (gfx1250) — matrix operand order repaired and checked by reading, not compiled, not run on hardware
//
#include <hip/hip_runtime.h>
#include <hip/hip_bf16.h>
#include <cstdint>

typedef __bf16 bf16;
typedef __attribute__((ext_vector_type(16))) __bf16 v16bf;
typedef __attribute__((ext_vector_type(8)))  float  v8f;

#define B_ 8
#define C_ 256
#define N_ 1024
#define HEADS 8
#define HDIM 32
#define SCALE 0.1767766952966369f  /* 1/sqrt(32) */

// ---------- 16-bit A-fragment K index used here (contiguous; B loads use the same order) ----------
// 16-bit A 16x32: elem e (0..15):
//   k = half*16 + e
__device__ __forceinline__ int a_kidx(int e, int lh) {
    return e + (lh << 4);
}

// ---------------- K0: f32 -> bf16 convert ----------------
__global__ void cvt_bf16_kernel(const float* __restrict__ src, bf16* __restrict__ dst, int n) {
    int i = blockIdx.x * blockDim.x + threadIdx.x;
    if (i < n) dst[i] = (bf16)src[i];
}

// ---------------- K1: per-pixel LayerNorm over C, write xnT [b][n][c] bf16 ----------------
__global__ void ln_kernel(const float* __restrict__ x, const float* __restrict__ gamma,
                          const float* __restrict__ beta, bf16* __restrict__ xnT) {
    int p = blockIdx.x;            // pixel index in [0, B*N)
    int b = p / N_, n = p % N_;
    int c = threadIdx.x;           // 256 threads
    float v = x[((size_t)b * C_ + c) * N_ + n];
    __shared__ float s1[C_], s2[C_];
    s1[c] = v; s2[c] = v * v;
    __syncthreads();
    for (int off = 128; off > 0; off >>= 1) {
        if (c < off) { s1[c] += s1[c + off]; s2[c] += s2[c + off]; }
        __syncthreads();
    }
    float mu  = s1[0] * (1.0f / C_);
    float var = s2[0] * (1.0f / C_) - mu * mu;
    float xn  = (v - mu) * rsqrtf(var + 1e-5f);
    xn = xn * gamma[c] + beta[c];
    xnT[((size_t)b * N_ + n) * C_ + c] = (bf16)xn;
}

// ---------------- K2: QKV GEMM, 16x64 output per wave (A reused 4x) ----------------
// grid (48 Mtiles, 4, B), block 128 (4 waves, each wave -> 4 consecutive N-tiles)
__global__ void qkv_kernel(const bf16* __restrict__ wq, const bf16* __restrict__ xnT,
                           const float* __restrict__ b_qkv,
                           bf16* __restrict__ qT, bf16* __restrict__ kT, bf16* __restrict__ vM) {
    int lane = threadIdx.x & 31, wv = threadIdx.x >> 5;
    int lh = lane >> 4, l16 = lane & 15;
    int mt  = blockIdx.x;                  // 0..47
    int nt0 = (blockIdx.y * 4 + wv) * 4;   // 0,4,...,60
    int b   = blockIdx.z;
    const bf16* xn = xnT + (size_t)b * N_ * C_;

    v8f acc0 = {}, acc1 = {}, acc2 = {}, acc3 = {};
    #pragma unroll
    for (int ks = 0; ks < 8; ++ks) {
        int k0 = ks * 32;
        const bf16* arow = wq + (size_t)(mt * 16 + l16) * C_ + k0;
        v16bf a;
        #pragma unroll
        for (int e = 0; e < 16; ++e) a[e] = arow[a_kidx(e, lh)];
        const bf16* bbase = xn + (size_t)l16 * C_ + k0 + (lh << 4);
        v16bf bf0 = *(const v16bf*)(bbase + (size_t)(nt0 + 0) * 16 * C_);
        v16bf bf1 = *(const v16bf*)(bbase + (size_t)(nt0 + 1) * 16 * C_);
        v16bf bf2 = *(const v16bf*)(bbase + (size_t)(nt0 + 2) * 16 * C_);
        v16bf bf3 = *(const v16bf*)(bbase + (size_t)(nt0 + 3) * 16 * C_);
        acc0 = __builtin_amdgcn_wmma_f32_16x16x32_bf16(false, a, false, bf0, (short)0, acc0, false, false);
        acc1 = __builtin_amdgcn_wmma_f32_16x16x32_bf16(false, a, false, bf1, (short)0, acc1, false, false);
        acc2 = __builtin_amdgcn_wmma_f32_16x16x32_bf16(false, a, false, bf2, (short)0, acc2, false, false);
        acc3 = __builtin_amdgcn_wmma_f32_16x16x32_bf16(false, a, false, bf3, (short)0, acc3, false, false);
    }
    v8f accs[4] = {acc0, acc1, acc2, acc3};
    #pragma unroll
    for (int r = 0; r < 8; ++r) {
        int o = mt * 16 + r + (lh << 3);
        float bias = b_qkv[o];
        int which = o >> 8, hh = (o >> 5) & 7, d = o & 31;
        size_t bh = (size_t)b * HEADS + hh;
        #pragma unroll
        for (int j = 0; j < 4; ++j) {
            int nPix = (nt0 + j) * 16 + l16;
            bf16 bv = (bf16)(accs[j][r] + bias);
            if (which == 0)      qT[(bh * N_ + nPix) * HDIM + d] = bv;
            else if (which == 1) kT[(bh * N_ + nPix) * HDIM + d] = bv;
            else                 vM[(bh * HDIM + d) * N_ + nPix] = bv;
        }
    }
}

// ---------------- K3: flash attention, one wave per 16-query tile ----------------
// grid (64 bh, 16), block 128 (4 waves, wave = query subtile)
__global__ void attn_kernel(const bf16* __restrict__ qT, const bf16* __restrict__ kT,
                            const bf16* __restrict__ vM, bf16* __restrict__ obuf) {
    int lane = threadIdx.x & 31, wv = threadIdx.x >> 5;
    int lh = lane >> 4, l16 = lane & 15;
    int bh = blockIdx.x;
    int qt = blockIdx.y * 4 + wv;        // 0..63
    const bf16* qp = qT + (size_t)bh * N_ * HDIM;   // [n][d]
    const bf16* kp = kT + (size_t)bh * N_ * HDIM;   // [m][d]
    const bf16* vp = vM + (size_t)bh * HDIM * N_;   // [d][m]

    // Q A-fragment (rows = 16 queries, K = d = 32), loaded once
    v16bf aq;
    {
        const bf16* qrow = qp + (size_t)(qt * 16 + l16) * HDIM;
        #pragma unroll
        for (int e = 0; e < 16; ++e) aq[e] = qrow[a_kidx(e, lh)];
    }

    float mrun[8], lrun[8];
    #pragma unroll
    for (int r = 0; r < 8; ++r) { mrun[r] = -1e30f; lrun[r] = 0.0f; }
    v8f o0 = {}, o1 = {};                 // d rows 0..15 / 16..31, cols = queries
    const v8f zero = {};

    __shared__ __align__(64) bf16 plds[4][16][32];  // per wave: P[q][m]
    __shared__ float fac[4][16], linv[4][16];

    for (int mb = 0; mb < 32; ++mb) {     // 32 key-blocks of 32
        int m0 = mb * 32;
        // S tiles: rows=queries, cols=keys; K = d = 32
        v8f s0, s1;
        {
            v16bf bk0 = *(const v16bf*)(kp + (size_t)(m0 + l16) * HDIM + (lh << 4));
            v16bf bk1 = *(const v16bf*)(kp + (size_t)(m0 + 16 + l16) * HDIM + (lh << 4));
            s0 = __builtin_amdgcn_wmma_f32_16x16x32_bf16(false, aq, false, bk0, (short)0, zero, false, false);
            s1 = __builtin_amdgcn_wmma_f32_16x16x32_bf16(false, aq, false, bk1, (short)0, zero, false, false);
        }
        // online softmax per query row (row = r + lh*8, spread across 16 lanes = keys)
        #pragma unroll
        for (int r = 0; r < 8; ++r) {
            float v0 = s0[r] * SCALE, v1 = s1[r] * SCALE;
            float mx = fmaxf(v0, v1);
            #pragma unroll
            for (int msk = 1; msk < 16; msk <<= 1) mx = fmaxf(mx, __shfl_xor(mx, msk, 32));
            float newm = fmaxf(mrun[r], mx);
            float f  = __expf(mrun[r] - newm);
            float p0 = __expf(v0 - newm), p1 = __expf(v1 - newm);
            float rs = p0 + p1;
            #pragma unroll
            for (int msk = 1; msk < 16; msk <<= 1) rs += __shfl_xor(rs, msk, 32);
            lrun[r] = lrun[r] * f + rs;
            mrun[r] = newm;
            int q = (lh << 3) + r;
            plds[wv][q][l16]      = (bf16)p0;
            plds[wv][q][16 + l16] = (bf16)p1;
            if (l16 == 0) fac[wv][q] = f;
        }
        __syncthreads();
        // rescale O (cols = query = l16)
        float fcol = fac[wv][l16];
        #pragma unroll
        for (int r = 0; r < 8; ++r) { o0[r] *= fcol; o1[r] *= fcol; }
        // O += V[d, m] * P^T[m, q]   (K = m = 32)
        v16bf bp = *(const v16bf*)(&plds[wv][l16][lh << 4]);
        {
            const bf16* vrow0 = vp + (size_t)l16 * N_ + m0;
            const bf16* vrow1 = vp + (size_t)(16 + l16) * N_ + m0;
            v16bf av0, av1;
            #pragma unroll
            for (int e = 0; e < 16; ++e) { int kk = a_kidx(e, lh); av0[e] = vrow0[kk]; av1[e] = vrow1[kk]; }
            o0 = __builtin_amdgcn_wmma_f32_16x16x32_bf16(false, av0, false, bp, (short)0, o0, false, false);
            o1 = __builtin_amdgcn_wmma_f32_16x16x32_bf16(false, av1, false, bp, (short)0, o1, false, false);
        }
        __syncthreads();
    }
    // finalize: broadcast 1/l per query and store O^T into obuf [b][n][h*32+d]
    if (l16 == 0) {
        #pragma unroll
        for (int r = 0; r < 8; ++r) linv[wv][(lh << 3) + r] = 1.0f / lrun[r];
    }
    __syncthreads();
    float il = linv[wv][l16];
    int b = bh >> 3, h = bh & 7;
    int nPix = qt * 16 + l16;
    bf16* orow = obuf + ((size_t)b * N_ + nPix) * C_ + h * HDIM;
    #pragma unroll
    for (int r = 0; r < 8; ++r) {
        int d = r + (lh << 3);
        orow[d]      = (bf16)(o0[r] * il);
        orow[16 + d] = (bf16)(o1[r] * il);
    }
}

// ---------------- K4: proj GEMM + bias + residual, 16x64 per wave, fp32 out ----------------
// grid (16 Mtiles, 4, B), block 128
__global__ void proj_kernel(const bf16* __restrict__ wp, const bf16* __restrict__ obuf,
                            const float* __restrict__ b_proj, const float* __restrict__ x,
                            float* __restrict__ out) {
    int lane = threadIdx.x & 31, wv = threadIdx.x >> 5;
    int lh = lane >> 4, l16 = lane & 15;
    int mt  = blockIdx.x;                  // 0..15
    int nt0 = (blockIdx.y * 4 + wv) * 4;   // 0,4,...,60
    int b   = blockIdx.z;
    const bf16* ob = obuf + (size_t)b * N_ * C_;

    v8f acc0 = {}, acc1 = {}, acc2 = {}, acc3 = {};
    #pragma unroll
    for (int ks = 0; ks < 8; ++ks) {
        int k0 = ks * 32;
        const bf16* arow = wp + (size_t)(mt * 16 + l16) * C_ + k0;
        v16bf a;
        #pragma unroll
        for (int e = 0; e < 16; ++e) a[e] = arow[a_kidx(e, lh)];
        const bf16* bbase = ob + (size_t)l16 * C_ + k0 + (lh << 4);
        v16bf bf0 = *(const v16bf*)(bbase + (size_t)(nt0 + 0) * 16 * C_);
        v16bf bf1 = *(const v16bf*)(bbase + (size_t)(nt0 + 1) * 16 * C_);
        v16bf bf2 = *(const v16bf*)(bbase + (size_t)(nt0 + 2) * 16 * C_);
        v16bf bf3 = *(const v16bf*)(bbase + (size_t)(nt0 + 3) * 16 * C_);
        acc0 = __builtin_amdgcn_wmma_f32_16x16x32_bf16(false, a, false, bf0, (short)0, acc0, false, false);
        acc1 = __builtin_amdgcn_wmma_f32_16x16x32_bf16(false, a, false, bf1, (short)0, acc1, false, false);
        acc2 = __builtin_amdgcn_wmma_f32_16x16x32_bf16(false, a, false, bf2, (short)0, acc2, false, false);
        acc3 = __builtin_amdgcn_wmma_f32_16x16x32_bf16(false, a, false, bf3, (short)0, acc3, false, false);
    }
    v8f accs[4] = {acc0, acc1, acc2, acc3};
    #pragma unroll
    for (int r = 0; r < 8; ++r) {
        int c = mt * 16 + r + (lh << 3);
        float bias = b_proj[c];
        #pragma unroll
        for (int j = 0; j < 4; ++j) {
            int nPix = (nt0 + j) * 16 + l16;
            size_t idx = ((size_t)b * C_ + c) * N_ + nPix;
            out[idx] = x[idx] + accs[j][r] + bias;
        }
    }
}

extern "C" void kernel_launch(void* const* d_in, const int* in_sizes, int n_in,
                              void* d_out, int out_size, void* d_ws, size_t ws_size,
                              hipStream_t stream) {
    const float* x      = (const float*)d_in[0];
    const float* gamma  = (const float*)d_in[1];
    const float* beta   = (const float*)d_in[2];
    const float* w_qkv  = (const float*)d_in[3];
    const float* b_qkv  = (const float*)d_in[4];
    const float* w_proj = (const float*)d_in[5];
    const float* b_proj = (const float*)d_in[6];
    float* out = (float*)d_out;

    size_t off = 0;
    auto alloc = [&](size_t bytes) -> void* {
        void* p = (char*)d_ws + off;
        off += (bytes + 255) & ~(size_t)255;
        return p;
    };
    bf16* wqkv_bf  = (bf16*)alloc((size_t)768 * 256 * 2);
    bf16* wproj_bf = (bf16*)alloc((size_t)256 * 256 * 2);
    bf16* xnT      = (bf16*)alloc((size_t)B_ * N_ * C_ * 2);
    bf16* qT       = (bf16*)alloc((size_t)B_ * HEADS * N_ * HDIM * 2);
    bf16* kT       = (bf16*)alloc((size_t)B_ * HEADS * N_ * HDIM * 2);
    bf16* vM       = (bf16*)alloc((size_t)B_ * HEADS * HDIM * N_ * 2);
    bf16* obuf     = (bf16*)alloc((size_t)B_ * N_ * C_ * 2);
    (void)ws_size;

    cvt_bf16_kernel<<<(768 * 256 + 255) / 256, 256, 0, stream>>>(w_qkv, wqkv_bf, 768 * 256);
    cvt_bf16_kernel<<<(256 * 256 + 255) / 256, 256, 0, stream>>>(w_proj, wproj_bf, 256 * 256);
    ln_kernel<<<B_ * N_, 256, 0, stream>>>(x, gamma, beta, xnT);
    qkv_kernel<<<dim3(48, 4, B_), 128, 0, stream>>>(wqkv_bf, xnT, b_qkv, qT, kT, vM);
    attn_kernel<<<dim3(B_ * HEADS, 16), 128, 0, stream>>>(qT, kT, vM, obuf);
    proj_kernel<<<dim3(16, 4, B_), 128, 0, stream>>>(wproj_bf, obuf, b_proj, x, out);
}
